// MultiheadAttention_21474836480255
// MI455X (gfx1250) — matrix operand order repaired and checked by reading, not compiled, not run on hardware
//
#include <hip/hip_runtime.h>

// MHA forward, gfx1250 (MI455X). All matmuls on v_wmma_f32_16x16x32_f16.
// d_out = [out: B*T*D f32][attn: B*H*T*T f32].
// Pipeline:
//   cvt x/w_qkv/w_proj -> f16 (once; kills redundant cvt VALU + halves read BW)
//   qkv_gemm (pure f16 WMMA)
//   fused flash-style attention: scores+softmax+context in one pass;
//     V chunks staged global->LDS with GLOBAL_LOAD_ASYNC_TO_LDS_B128 (ASYNCcnt),
//     probability tiles bounced through LDS into the attn@V WMMA,
//     one L2-hot normalize sweep over the just-written attn strip
//   out projection (pure f16 WMMA)

typedef __attribute__((ext_vector_type(16))) _Float16 v16h;
typedef __attribute__((ext_vector_type(4)))  _Float16 v4h;
typedef __attribute__((ext_vector_type(8)))  float    v8f;

#define Bsz   2
#define Tsz   2048
#define Dm    1024
#define Hn    16
#define Dh    64
#define E3    3072
#define SCALE 0.125f   // 1/sqrt(64)

static __device__ __forceinline__ v8f wmma32(v16h a, v16h b, v8f c) {
  // 8 args: (neg_a, A, neg_b, B, c_mod, C, reuse_a, reuse_b)
  return __builtin_amdgcn_wmma_f32_16x16x32_f16(false, a, false, b, (short)0, c, false, false);
}

// ---- fragment loaders (contiguous k order, same in A and B, wave32) -------------------------
// A 16x32 (MxK): lanes 0-15 row=lane, K 0-15; lanes 16-31 K 16-31
static __device__ __forceinline__ v16h ldA_f16(const _Float16* p, int ld, int lane) {
  const int row = lane & 15;
  const int kb  = (lane & 16) ? 16 : 0;
  const _Float16* q = p + row * ld + kb;
  v16h a;
#pragma unroll
  for (int h = 0; h < 16; ++h) a[h] = q[h];
  return a;
}
// B 32x16 (KxN), source stored N-major (element (k,n) at p[n*ld+k]), i.e. a
// row-major weight W[n][k] used as B=W^T. lanes 0-15: K 0-15; lanes 16-31: K 16-31.
static __device__ __forceinline__ v16h ldB_nmaj_f16(const _Float16* p, int ld, int lane) {
  const int col = lane & 15;
  const int kb  = (lane & 16) ? 16 : 0;
  const _Float16* q = p + col * ld + kb;
  v16h b;
#pragma unroll
  for (int h = 0; h < 16; ++h) b[h] = q[h];
  return b;
}
// B 32x16 (KxN), source stored K-major (element (k,n) at p[k*ld+n]) — V from LDS.
static __device__ __forceinline__ v16h ldB_kmaj_f16(const _Float16* p, int ld, int lane) {
  const int col = lane & 15;
  const int kb  = (lane & 16) ? 16 : 0;
  v16h b;
#pragma unroll
  for (int h = 0; h < 16; ++h) b[h] = p[(kb + h) * ld + col];
  return b;
}

// ---- kernel 0: one-time f32 -> f16 conversion ------------------------------
__global__ void cvt_f32_to_f16(const float* __restrict__ src,
                               _Float16* __restrict__ dst, int n4) {
  const int i = blockIdx.x * blockDim.x + threadIdx.x;
  if (i >= n4) return;
  const float4 v = ((const float4*)src)[i];
  v4h h;
  h[0] = (_Float16)v.x; h[1] = (_Float16)v.y;
  h[2] = (_Float16)v.z; h[3] = (_Float16)v.w;
  ((v4h*)dst)[i] = h;
}

// ---- kernel 1: qkv = x @ w_qkv^T (all f16), scatter into per-head Q/K/V ----
__global__ void mha_qkv_gemm(const _Float16* __restrict__ x,
                             const _Float16* __restrict__ w_qkv,
                             _Float16* __restrict__ qkv) {
  const int lane = threadIdx.x;
  const int mt = blockIdx.x;   // 256 M-tiles (M = B*T = 4096)
  const int nt = blockIdx.y;   // 48 strips of 64 over E3=3072
  const _Float16* xrow = x     + (size_t)(mt * 16) * Dm;
  const _Float16* wrow = w_qkv + (size_t)(nt * 64) * Dm;
  v8f acc[4] = {};
  for (int k0 = 0; k0 < Dm; k0 += 32) {
    __builtin_prefetch(xrow + (lane & 15) * Dm + k0 + 128, 0, 0);
    v16h a = ldA_f16(xrow + k0, Dm, lane);
#pragma unroll
    for (int j = 0; j < 4; ++j) {
      v16h b = ldB_nmaj_f16(wrow + (size_t)(j * 16) * Dm + k0, Dm, lane);
      acc[j] = wmma32(a, b, acc[j]);
    }
  }
  const int col = lane & 15;
  const int rb  = (lane & 16) ? 8 : 0;
#pragma unroll
  for (int j = 0; j < 4; ++j) {
#pragma unroll
    for (int r = 0; r < 8; ++r) {
      const int m = mt * 16 + rb + r;
      const int e = nt * 64 + j * 16 + col;
      const int which = e >> 10, rem = e & 1023;
      const int h = rem >> 6, dh = rem & 63;
      const int bb = m >> 11, t = m & (Tsz - 1);
      const size_t idx = ((size_t)(which * (Bsz * Hn) + bb * Hn + h) * Tsz + t) * Dh + dh;
      qkv[idx] = (_Float16)acc[j][r];
    }
  }
}

// ---- kernel 2: fused scores + softmax + context (one wave per (b,h,qtile)) -
__global__ void mha_fused_attn(const _Float16* __restrict__ qkv,
                               float* __restrict__ attn,
                               _Float16* __restrict__ ctx) {
  __shared__ _Float16 pbuf[16 * 32];    // probability tile, C-layout -> A-layout bounce
  __shared__ _Float16 vbuf[32 * Dh];    // V chunk, filled by async global->LDS DMA
  const int lane = threadIdx.x;
  const int qt = blockIdx.x;            // 128 query tiles
  const int bh = blockIdx.y;            // 32 (b,h) pairs
  const _Float16* Q = qkv + (size_t)bh * Tsz * Dh;
  const _Float16* K = qkv + (size_t)(Bsz * Hn + bh) * Tsz * Dh;
  const _Float16* V = qkv + (size_t)(2 * Bsz * Hn + bh) * Tsz * Dh;
  const v16h a0 = ldA_f16(Q + (size_t)(qt * 16) * Dh +  0, Dh, lane);
  const v16h a1 = ldA_f16(Q + (size_t)(qt * 16) * Dh + 32, Dh, lane);
  const int col = lane & 15;
  const int rb  = (lane & 16) ? 8 : 0;
  float* arow = attn + ((size_t)bh * Tsz + (size_t)(qt * 16)) * Tsz;

  // low 32 bits of the generic LDS address == LDS byte offset (aperture rule)
  const unsigned vbuf_lds = (unsigned)(uintptr_t)&vbuf[0];

  float lsum[8] = {};                   // per-lane partial row sums
  v8f acc[4] = {};                      // context accumulators (16 x 64)

  const int nchunk = (qt + 2) / 2;      // 32-wide key chunks covering keys <= qt*16+15
  for (int s = 0; s < nchunk; ++s) {
    // kick off async DMA of the 4KB V[32 x 64] chunk into LDS (ASYNCcnt)
    {
      const unsigned long long gbase =
          (unsigned long long)(uintptr_t)(V + (size_t)(s * 32) * Dh);
      unsigned goff = (unsigned)lane * 16u;
#pragma unroll
      for (int i = 0; i < 8; ++i) {
        const unsigned ldsoff = vbuf_lds + goff;
        asm volatile("global_load_async_to_lds_b128 %0, %1, %2"
                     :: "v"(ldsoff), "v"(goff), "s"(gbase) : "memory");
        goff += 512u;   // 32 lanes * 16B
      }
    }
    // meanwhile: two 16x16 score tiles (keys 32s..32s+31)
#pragma unroll
    for (int half = 0; half < 2; ++half) {
      const int jt = 2 * s + half;
      v8f c = {};
      if (jt <= qt) {
        v16h b0 = ldB_nmaj_f16(K + (size_t)(jt * 16) * Dh +  0, Dh, lane);
        v16h b1 = ldB_nmaj_f16(K + (size_t)(jt * 16) * Dh + 32, Dh, lane);
        c = wmma32(a0, b0, c);
        c = wmma32(a1, b1, c);
      }
#pragma unroll
      for (int r = 0; r < 8; ++r) {
        const int qi = qt * 16 + rb + r;
        const int ki = jt * 16 + col;
        float p = 0.0f;
        if (jt <= qt && ki <= qi) p = __expf(c[r] * SCALE);
        lsum[r] += p;
        arow[(size_t)(rb + r) * Tsz + ki] = p;               // unnormalized prob
        pbuf[(rb + r) * 32 + half * 16 + col] = (_Float16)p; // A-frag staging
      }
    }
    // probabilities as A fragment (wave-local DS ops are in order: no barrier)
    v16h pa = ldA_f16(pbuf, 32, lane);
    // V DMA must have landed before we read vbuf
    asm volatile("s_wait_asynccnt 0x0" ::: "memory");
#pragma unroll
    for (int j = 0; j < 4; ++j) {
      v16h b = ldB_kmaj_f16(vbuf + j * 16, Dh, lane);
      acc[j] = wmma32(pa, b, acc[j]);
    }
  }

  // zero-fill fully masked key tiles beyond the processed range (write-only)
  for (int jt = 2 * nchunk; jt < Tsz / 16; ++jt)
#pragma unroll
    for (int r = 0; r < 8; ++r)
      __builtin_nontemporal_store(0.0f, &arow[(size_t)(rb + r) * Tsz + jt * 16 + col]);

  // complete row sums across each 16-lane half (rows rb..rb+7 per half)
  float inv[8];
#pragma unroll
  for (int r = 0; r < 8; ++r) {
    float t = lsum[r];
#pragma unroll
    for (int off = 8; off; off >>= 1) t += __shfl_xor(t, off, 32);
    inv[r] = 1.0f / t;
  }

  // normalize stored probabilities (just-written strip: L2-hot)
  const int kmax = (qt + 1) * 16;
#pragma unroll
  for (int r = 0; r < 8; ++r)
    for (int k = col; k < kmax; k += 16)
      arow[(size_t)(rb + r) * Tsz + k] *= inv[r];

  // normalize and store context
  const int bb = bh >> 4, h = bh & 15;
#pragma unroll
  for (int j = 0; j < 4; ++j)
#pragma unroll
    for (int r = 0; r < 8; ++r) {
      const int t = qt * 16 + rb + r;
      ctx[(size_t)(bb * Tsz + t) * Dm + h * Dh + j * 16 + col] =
          (_Float16)(acc[j][r] * inv[r]);
    }
}

// ---- kernel 3: out = context @ w_proj^T (all f16) --------------------------
__global__ void mha_proj(const _Float16* __restrict__ ctx,
                         const _Float16* __restrict__ w_proj,
                         float* __restrict__ out) {
  const int lane = threadIdx.x;
  const int mt = blockIdx.x;   // 256
  const int nt = blockIdx.y;   // 16 strips of 64 over 1024
  const _Float16* crow = ctx    + (size_t)(mt * 16) * Dm;
  const _Float16* wrow = w_proj + (size_t)(nt * 64) * Dm;
  v8f acc[4] = {};
  for (int k0 = 0; k0 < Dm; k0 += 32) {
    __builtin_prefetch(wrow + (lane & 15) * Dm + k0 + 128, 0, 0);
    v16h a = ldA_f16(crow + k0, Dm, lane);
#pragma unroll
    for (int j = 0; j < 4; ++j) {
      v16h b = ldB_nmaj_f16(wrow + (size_t)(j * 16) * Dm + k0, Dm, lane);
      acc[j] = wmma32(a, b, acc[j]);
    }
  }
  const int col = lane & 15;
  const int rb  = (lane & 16) ? 8 : 0;
#pragma unroll
  for (int j = 0; j < 4; ++j)
#pragma unroll
    for (int r = 0; r < 8; ++r) {
      const int m = mt * 16 + rb + r;
      const int n = nt * 64 + j * 16 + col;
      __builtin_nontemporal_store(acc[j][r], &out[(size_t)m * Dm + n]);
    }
}

extern "C" void kernel_launch(void* const* d_in, const int* in_sizes, int n_in,
                              void* d_out, int out_size, void* d_ws, size_t ws_size,
                              hipStream_t stream) {
  (void)in_sizes; (void)n_in; (void)out_size; (void)ws_size;
  const float* x      = (const float*)d_in[0];
  const float* w_qkv  = (const float*)d_in[1];
  const float* w_proj = (const float*)d_in[2];

  float* out  = (float*)d_out;
  float* attn = out + (size_t)Bsz * Tsz * Dm;            // tuple order: (out, attn)

  // workspace (f16 elements): qkv | ctx | x16 | w_qkv16 | w_proj16  (~50 MB)
  _Float16* qkv  = (_Float16*)d_ws;                      // [3][B][H][T][Dh]
  _Float16* ctx  = qkv + (size_t)3 * Bsz * Hn * Tsz * Dh;
  _Float16* x16  = ctx + (size_t)Bsz * Tsz * Dm;         // [B*T][D]
  _Float16* wq16 = x16 + (size_t)Bsz * Tsz * Dm;         // [3D][D]
  _Float16* wp16 = wq16 + (size_t)E3 * Dm;               // [D][D]

  const int n4_x  = (Bsz * Tsz * Dm) / 4;                // 1M float4
  const int n4_wq = (E3 * Dm) / 4;                       // 768K
  const int n4_wp = (Dm * Dm) / 4;                       // 256K
  cvt_f32_to_f16<<<(n4_x  + 255) / 256, 256, 0, stream>>>(x,      x16, n4_x);
  cvt_f32_to_f16<<<(n4_wq + 255) / 256, 256, 0, stream>>>(w_qkv,  wq16, n4_wq);
  cvt_f32_to_f16<<<(n4_wp + 255) / 256, 256, 0, stream>>>(w_proj, wp16, n4_wp);

  mha_qkv_gemm  <<<dim3(256, 48), 32, 0, stream>>>(x16, wq16, qkv);
  mha_fused_attn<<<dim3(Tsz / 16, Bsz * Hn), 32, 0, stream>>>(qkv, attn, ctx);
  mha_proj      <<<dim3(256, Dm / 64), 32, 0, stream>>>(ctx, wp16, out);
}
